// CondConv2d_76476187672694
// MI455X (gfx1250) — compile-verified, matrix operand order checked by reading, not run on hardware
//
#include <hip/hip_runtime.h>
#include <hip/hip_bf16.h>
#include <math.h>

// ---------------------------------------------------------------------------
// CondConv2d: B=32, CIN=256, COUT=256, H=W=56, KH=KW=3, E=8
//   1) routing  = sigmoid(avgpool(x) @ route_w + route_b)        (B,E)
//   2) w_mix[b] = routing[b] @ expert_weights   -> bf16          (B,COUT,CIN*9)
//   3) out[b]   = W_b (256 x 2304) @ im2col(x_b) (2304 x 3136)   WMMA bf16
//      A-tile staged to LDS by the Tensor Data Mover (tensor_load_to_lds),
//      B-tile staged manually (needs fp32->bf16 cvt + conv zero-padding).
// ---------------------------------------------------------------------------

#define BATCH 32
#define CIN   256
#define COUT  256
#define HW    3136      // 56*56
#define HDIM  56
#define KDIM  2304      // CIN*9
#define NEXP  8
#define WELEMS 589824   // COUT*CIN*9

typedef __attribute__((ext_vector_type(16))) __bf16 v16bf;
typedef __attribute__((ext_vector_type(8)))  float  v8f;

#if __has_builtin(__builtin_amdgcn_tensor_load_to_lds) && \
    __has_builtin(__builtin_amdgcn_s_wait_tensorcnt)
#define USE_TDM 1
typedef __attribute__((ext_vector_type(4))) unsigned int v4u;
typedef __attribute__((ext_vector_type(8))) int          v8i_;
typedef __attribute__((ext_vector_type(4))) int          v4i_;
#endif

static __device__ inline __bf16 f2bf(float f) {
    union { float f; unsigned u; } uf; uf.f = f;
    unsigned r = uf.u + 0x7FFFu + ((uf.u >> 16) & 1u);   // round-to-nearest-even
    unsigned short h = (unsigned short)(r >> 16);
    __bf16 out; __builtin_memcpy(&out, &h, 2);
    return out;
}

// ---------------------------------------------------------------------------
// Kernel 1: routing = sigmoid(mean_hw(x) @ route_w + route_b)
// ---------------------------------------------------------------------------
__global__ void routing_kernel(const float* __restrict__ x,
                               const float* __restrict__ route_w,
                               const float* __restrict__ route_b,
                               float* __restrict__ routing) {
    __shared__ float pooled_s[CIN];
    const int b    = blockIdx.x;
    const int tid  = threadIdx.x;
    const int wid  = tid >> 5;      // wave32
    const int lane = tid & 31;

    for (int c = wid; c < CIN; c += 8) {
        const float* p = x + ((size_t)b * CIN + c) * HW;
        float s = 0.f;
        for (int i = lane; i < HW; i += 32) s += p[i];
        #pragma unroll
        for (int off = 16; off > 0; off >>= 1) s += __shfl_down(s, off, 32);
        if (lane == 0) pooled_s[c] = s * (1.0f / (float)HW);
    }
    __syncthreads();

    if (tid < NEXP) {
        float acc = route_b[tid];
        for (int c = 0; c < CIN; ++c) acc += pooled_s[c] * route_w[c * NEXP + tid];
        routing[b * NEXP + tid] = 1.0f / (1.0f + expf(-acc));
    }
}

// ---------------------------------------------------------------------------
// Kernel 2: w_mix[b][i] = sum_e routing[b][e] * expert_w[e][i]  (bf16 out)
// ---------------------------------------------------------------------------
__global__ void mix_kernel(const float* __restrict__ expert_w,
                           const float* __restrict__ routing,
                           __bf16* __restrict__ wmix) {
    const int b = blockIdx.y;
    float r[NEXP];
    #pragma unroll
    for (int e = 0; e < NEXP; ++e) r[e] = routing[b * NEXP + e];

    const int i = (blockIdx.x * blockDim.x + threadIdx.x) * 4;
    if (i >= WELEMS) return;

    float4 acc = make_float4(0.f, 0.f, 0.f, 0.f);
    #pragma unroll
    for (int e = 0; e < NEXP; ++e) {
        const float4 wv = *(const float4*)(expert_w + (size_t)e * WELEMS + i);
        acc.x += r[e] * wv.x; acc.y += r[e] * wv.y;
        acc.z += r[e] * wv.z; acc.w += r[e] * wv.w;
    }
    __bf16 o[4] = { f2bf(acc.x), f2bf(acc.y), f2bf(acc.z), f2bf(acc.w) };
    unsigned long long pk; __builtin_memcpy(&pk, o, 8);
    *(unsigned long long*)(wmix + (size_t)b * WELEMS + i) = pk;   // 8B store
}

// ---------------------------------------------------------------------------
// Kernel 3: per-sample implicit-GEMM conv via v_wmma_f32_16x16x32_bf16
//   block = 256 threads (8 wave32s), tile BM=128 x BN=128, BK=32
//   A tile (weights, bf16) staged by TDM: one tensor_load_to_lds per K-step,
//   issued by wave 0; completion via s_wait_tensorcnt + workgroup barrier.
// ---------------------------------------------------------------------------
__global__ void __launch_bounds__(256)
condconv_wmma_kernel(const float* __restrict__ x,
                     const __bf16* __restrict__ wmix,
                     float* __restrict__ out) {
    __shared__ __bf16 ldsA[128 * 32];   // 8 KB  (BM x BK), [row][k], 64B rows
    __shared__ __bf16 ldsB[128 * 32];   // 8 KB  (BN x BK), [col][k]

    const int tid  = threadIdx.x;
    const int lane = tid & 31;
    const int wid  = tid >> 5;
    const int wm   = wid & 3;           // 0..3 -> 32-row strip
    const int wn   = wid >> 2;          // 0..1 -> 64-col strip

    const int b       = blockIdx.z;
    const int m_block = blockIdx.y * 128;       // COUT tile base
    const int n_block = blockIdx.x * 128;       // HW   tile base

    const __bf16* Wb   = wmix + (size_t)b * WELEMS;            // (256 x 2304) bf16
    const float*  xb   = x    + (size_t)b * CIN * HW;
    float*        outb = out  + (size_t)b * COUT * HW;

    v8f acc[2][4];
    {
        v8f z = {0.f,0.f,0.f,0.f,0.f,0.f,0.f,0.f};
        #pragma unroll
        for (int i = 0; i < 2; ++i)
            #pragma unroll
            for (int j = 0; j < 4; ++j) acc[i][j] = z;
    }

#if USE_TDM
    // D# group 1 (constant): data_size=2B, tensor 2304x256, tile 32x128,
    // dim0 stride = 2304 elements. Fields per CDNA5 ISA.
    const unsigned ldsA_base = (unsigned)(uintptr_t)&ldsA[0];
    const v8i_ g1c = { (int)0x00010000u,          // data_size=1 (2B), mask=0
                       (int)(KDIM << 16),         // tensor_dim0 = 2304 (low 16 in [31:16])
                       (int)(COUT << 16),         // tensor_dim1 = 256
                       (int)(32   << 16),         // tile_dim0   = 32
                       128,                       // tile_dim1   = 128 (tile_dim2=0)
                       KDIM,                      // tensor_dim0_stride = 2304
                       0, 0 };
    const v4i_ gz = { 0, 0, 0, 0 };               // groups 2/3: 2-D tensor
#else
    const int a_row  = tid >> 1;        // 0..127
    const int a_half = tid & 1;         // 0..1 (16 bf16 each)
#endif

    // B-tile loader mapping: tid -> (kk, 16-column group)
    const int b_kk   = tid >> 3;        // 0..31
    const int b_g    = tid & 7;         // 0..7 -> 16 columns each

    for (int k0 = 0; k0 < KDIM; k0 += 32) {
        __syncthreads();   // all waves done reading previous tiles

        // ---- stage A tile: 128 rows x 32 k of mixed weights (bf16)
#if USE_TDM
        if (wid == 0) {
            const unsigned long long ga =
                (unsigned long long)(uintptr_t)(Wb + (size_t)m_block * KDIM + k0);
            const v4u g0 = { 1u,                                    // count=1
                             ldsA_base,                             // lds_addr
                             (unsigned)ga,                          // global_addr[31:0]
                             ((unsigned)(ga >> 32) & 0x01FFFFFFu)   // global_addr[56:32]
                               | 0x80000000u };                     // type=2
#if defined(__clang_major__) && (__clang_major__ >= 23)
            __builtin_amdgcn_tensor_load_to_lds(g0, g1c, gz, gz,
                                                (v8i_){0,0,0,0,0,0,0,0}, 0);
#else
            __builtin_amdgcn_tensor_load_to_lds(g0, g1c, gz, gz, 0);
#endif
        }
#else
        {
            const __bf16* src = Wb + (size_t)(m_block + a_row) * KDIM + k0 + a_half * 16;
            *(v16bf*)&ldsA[a_row * 32 + a_half * 16] = *(const v16bf*)src;
        }
#endif

        // ---- stage B tile: im2col(x) 32 k-rows x 128 columns, stored [n][k]
        {
            const int k  = k0 + b_kk;
            const int ci = k / 9;
            const int rs = k - ci * 9;
            const int r  = rs / 3;
            const int s  = rs - r * 3;
            const float* xc = xb + (size_t)ci * HW;
            if (k0 + 32 < KDIM)   // warm L2 for the next channel slab
                __builtin_prefetch(xc + HW, 0, 1);
            const int nb = n_block + b_g * 16;
            #pragma unroll 4
            for (int c = 0; c < 16; ++c) {
                const int n = nb + c;
                float v = 0.f;
                if (n < HW) {
                    const int oh = n / HDIM;
                    const int ow = n - oh * HDIM;
                    const int ih = oh + r - 1;
                    const int iw = ow + s - 1;
                    if ((unsigned)ih < (unsigned)HDIM && (unsigned)iw < (unsigned)HDIM)
                        v = xc[ih * HDIM + iw];
                }
                ldsB[(b_g * 16 + c) * 32 + b_kk] = f2bf(v);
            }
        }

#if USE_TDM
        if (wid == 0) __builtin_amdgcn_s_wait_tensorcnt(0);  // TDM tile landed
#endif
        __syncthreads();   // publish ldsA (TDM) + ldsB to all waves

        // ---- fragments: one aligned 32B LDS read per fragment
        v16bf afrag[2], bfrag[4];
        const int frow = lane & 15;
        const int half = lane >> 4;         // K 0..15 vs 16..31 (contiguous k order, same for A and B)
        #pragma unroll
        for (int i = 0; i < 2; ++i)
            afrag[i] = *(const v16bf*)&ldsA[(wm * 32 + i * 16 + frow) * 32 + half * 16];
        #pragma unroll
        for (int j = 0; j < 4; ++j)
            bfrag[j] = *(const v16bf*)&ldsB[(wn * 64 + j * 16 + frow) * 32 + half * 16];

        // ---- 8 WMMAs per wave per K-step (EXEC all ones here)
        #pragma unroll
        for (int i = 0; i < 2; ++i)
            #pragma unroll
            for (int j = 0; j < 4; ++j)
                acc[i][j] = __builtin_amdgcn_wmma_f32_16x16x32_bf16(
                    false, afrag[i], false, bfrag[j],
                    (short)0, acc[i][j], false, false);
    }

    // ---- epilogue: D layout = VGPR r: lanes 0-15 M=r, lanes 16-31 M=r+8
    #pragma unroll
    for (int i = 0; i < 2; ++i) {
        const int m_base = m_block + wm * 32 + i * 16 + (lane >> 4) * 8;
        #pragma unroll
        for (int j = 0; j < 4; ++j) {
            const int n = n_block + wn * 64 + j * 16 + (lane & 15);
            if (n < HW) {
                #pragma unroll
                for (int r = 0; r < 8; ++r)
                    outb[(size_t)(m_base + r) * HW + n] = acc[i][j][r];
            }
        }
    }
}

// ---------------------------------------------------------------------------
extern "C" void kernel_launch(void* const* d_in, const int* in_sizes, int n_in,
                              void* d_out, int out_size, void* d_ws, size_t ws_size,
                              hipStream_t stream) {
    const float* x        = (const float*)d_in[0];   // (32,256,56,56)
    const float* expert_w = (const float*)d_in[1];   // (8, 589824)
    const float* route_w  = (const float*)d_in[2];   // (256, 8)
    const float* route_b  = (const float*)d_in[3];   // (8,)
    float* out = (float*)d_out;                      // (32,256,56,56) fp32

    float*  routing = (float*)d_ws;                        // 256 floats
    __bf16* wmix    = (__bf16*)((char*)d_ws + 1024);       // 32*589824 bf16 = 37.75MB

    // 1) routing gates
    routing_kernel<<<dim3(BATCH), dim3(256), 0, stream>>>(x, route_w, route_b, routing);

    // 2) mix expert kernels -> bf16 per-sample weights
    {
        dim3 grid((WELEMS / 4 + 255) / 256, BATCH);
        mix_kernel<<<grid, dim3(256), 0, stream>>>(expert_w, routing, wmix);
    }

    // 3) WMMA implicit-GEMM conv: grid = (ceil(3136/128)=25, 256/128=2, 32)
    {
        dim3 grid((HW + 127) / 128, COUT / 128, BATCH);
        condconv_wmma_kernel<<<grid, dim3(256), 0, stream>>>(x, wmix, out);
    }
}
